// InfoNCELoss_60739427500160
// MI455X (gfx1250) — matrix operand order repaired and checked by reading, compiled, not run on hardware
//
#include <hip/hip_runtime.h>
#include <hip/hip_bf16.h>

typedef __attribute__((ext_vector_type(4)))  _Float16 v4h;
typedef __attribute__((ext_vector_type(8)))  _Float16 v8h;
typedef __attribute__((ext_vector_type(16))) _Float16 v16h;
typedef __attribute__((ext_vector_type(8)))  float    v8f;

#define NROWS 16384
#define KDIM  128
#define NTILES (NROWS / 16)   // 1024
#define TEMP_INV 2.0f         // 1 / 0.5

union H16 { v16h v; v8h h[2]; };

// ---------------------------------------------------------------------------
// Kernel 1: L2-normalize each row (fp32 -> f16), zero the global accumulator.
// ---------------------------------------------------------------------------
__global__ void infonce_norm_kernel(const float* __restrict__ z,
                                    _Float16* __restrict__ zn,
                                    float* __restrict__ acc) {
  if (blockIdx.x == 0 && threadIdx.x == 0) *acc = 0.0f;
  const int row  = blockIdx.x * (blockDim.x >> 5) + (threadIdx.x >> 5);
  const int lane = threadIdx.x & 31;
  const float4 v = ((const float4*)(z + (size_t)row * KDIM))[lane];
  float ss = v.x * v.x + v.y * v.y + v.z * v.z + v.w * v.w;
#pragma unroll
  for (int off = 16; off > 0; off >>= 1) ss += __shfl_xor(ss, off, 32);
  const float inv = 1.0f / fmaxf(sqrtf(ss), 1e-12f);
  v4h h;
  h[0] = (_Float16)(v.x * inv);
  h[1] = (_Float16)(v.y * inv);
  h[2] = (_Float16)(v.z * inv);
  h[3] = (_Float16)(v.w * inv);
  *(v4h*)(zn + (size_t)row * KDIM + lane * 4) = h;
}

// --- helpers ---------------------------------------------------------------

// B panel (32x16 f16 per k-chunk): lane holds col n=lane&15,
// halves 0-7 -> K = 8*hi + 0..7, halves 8-15 -> K = 16 + 8*hi + 0..7.
__device__ __forceinline__ void load_b(const _Float16* __restrict__ zn,
                                       int nt, int lh, int hi, H16 b[4]) {
  const _Float16* brow = zn + (size_t)(16 * nt + lh) * KDIM;
#pragma unroll
  for (int kb = 0; kb < 4; ++kb) {
    b[kb].h[0] = *(const v8h*)(brow + kb * 32 + hi * 8);
    b[kb].h[1] = *(const v8h*)(brow + kb * 32 + 16 + hi * 8);
  }
}

// A panel (16x32 f16 per k-chunk): lane holds row m=lane&15,
// halves 0-7 -> K = 8*hi + 0..7, halves 8-15 -> K = 16 + 8*hi + 0..7.
__device__ __forceinline__ void load_a(const _Float16* __restrict__ zn,
                                       int rowbase, int lh, int hi, H16 a[4]) {
  const _Float16* arow = zn + (size_t)(rowbase + lh) * KDIM;
#pragma unroll
  for (int kb = 0; kb < 4; ++kb) {
    a[kb].h[0] = *(const v8h*)(arow + kb * 32 + hi * 8);
    a[kb].h[1] = *(const v8h*)(arow + kb * 32 + 16 + hi * 8);
  }
}

__device__ __forceinline__ v8f tile_c(const H16 a[4], const H16 b[4]) {
  v8f c = {};
#pragma unroll
  for (int kb = 0; kb < 4; ++kb)
    c = __builtin_amdgcn_wmma_f32_16x16x32_f16(
        false, a[kb].v, false, b[kb].v, (short)0, c, false, false);
  return c;
}

__device__ __forceinline__ void acc_exp(const v8f& c, float sum[8]) {
#pragma unroll
  for (int r = 0; r < 8; ++r) sum[r] += __expf(TEMP_INV * c[r]);
}

// Diagonal-tile correction: exact-cancel the self-similarity exp that the
// uniform main loop added, and record the positive pair (i^1 is in-tile).
__device__ __forceinline__ void diag_fix(const v8f& c, float sum[8],
                                         float* pospair, int lh, int hi) {
#pragma unroll
  for (int r = 0; r < 8; ++r) {
    const int row = r + hi * 8;                  // local row 0..15
    const float v = TEMP_INV * c[r];
    if (lh == row) sum[r] -= __expf(v);          // bitwise-identical cancel
    if (lh == (row ^ 1)) pospair[row] = v;
  }
}

// ---------------------------------------------------------------------------
// Kernel 2: fused sim = (Zn Zn^T)/T + per-row sum(exp) + pos-pair extraction.
// Block (8 waves) owns a 32-row slab = two 16-row A panels; every B tile
// loaded from L2 feeds 8 WMMAs (2x reuse). Waves stripe over 1024 N-tiles
// with a register double-buffered B panel. Main loop is fully uniform; the
// two diagonal tiles are recomputed once by their owning waves afterwards.
// ---------------------------------------------------------------------------
__global__ void __launch_bounds__(256)
infonce_sim_kernel(const _Float16* __restrict__ zn, float* __restrict__ acc) {
  __shared__ float rowsum[32];
  __shared__ float pospair[32];

  const int mb   = blockIdx.x;        // 32-row slab index
  const int tid  = threadIdx.x;
  const int wave = tid >> 5;          // 0..7
  const int lane = tid & 31;
  const int lh   = lane & 15;
  const int hi   = lane >> 4;

  if (tid < 32) { rowsum[tid] = 0.0f; pospair[tid] = 0.0f; }
  __syncthreads();

  H16 a0[4], a1[4];
  load_a(zn, 32 * mb, lh, hi, a0);
  load_a(zn, 32 * mb + 16, lh, hi, a1);

  float sum0[8], sum1[8];
#pragma unroll
  for (int r = 0; r < 8; ++r) { sum0[r] = 0.0f; sum1[r] = 0.0f; }

  // --- uniform main sweep, software-pipelined (2 N-tiles/iter) ------------
  H16 b0[4], b1[4];
  load_b(zn, wave, lh, hi, b0);
  int nt = wave;
#pragma unroll 1
  for (int it = 0; it < NTILES / 16; ++it) {     // 64 iters, 2 tiles each
    load_b(zn, nt + 8, lh, hi, b1);              // prefetch tile nt+8
    acc_exp(tile_c(a0, b0), sum0);               // tile nt vs slab 0
    acc_exp(tile_c(a1, b0), sum1);               // tile nt vs slab 1
    int pf = nt + 16;
    if (pf >= NTILES) pf = wave;                 // harmless wrap on last iter
    load_b(zn, pf, lh, hi, b0);                  // prefetch tile nt+16
    acc_exp(tile_c(a0, b1), sum0);               // tile nt+8 vs slab 0
    acc_exp(tile_c(a1, b1), sum1);               // tile nt+8 vs slab 1
    nt += 16;
  }

  // --- diagonal corrections (two tiles, one owning wave each) -------------
  const int t0 = 2 * mb;                         // slab-0 diagonal N-tile
  const int t1 = 2 * mb + 1;                     // slab-1 diagonal N-tile
  if ((t0 & 7) == wave) {
    H16 bd[4];
    load_b(zn, t0, lh, hi, bd);
    diag_fix(tile_c(a0, bd), sum0, pospair, lh, hi);
  }
  if ((t1 & 7) == wave) {
    H16 bd[4];
    load_b(zn, t1, lh, hi, bd);
    diag_fix(tile_c(a1, bd), sum1, pospair + 16, lh, hi);
  }

  // --- combine per-lane partials across waves/lanes -----------------------
#pragma unroll
  for (int r = 0; r < 8; ++r) {
    atomicAdd(&rowsum[r + hi * 8], sum0[r]);
    atomicAdd(&rowsum[16 + r + hi * 8], sum1[r]);
  }
  __syncthreads();

  if (tid == 0) {
    float t = 0.0f;
#pragma unroll
    for (int r = 0; r < 32; ++r) t += __logf(rowsum[r]) - pospair[r];
    atomicAdd(acc, t);
  }
}

// ---------------------------------------------------------------------------
// Kernel 3: mean over the 16384 rows.
// ---------------------------------------------------------------------------
__global__ void infonce_final_kernel(const float* __restrict__ acc,
                                     float* __restrict__ out) {
  if (threadIdx.x == 0 && blockIdx.x == 0)
    out[0] = acc[0] * (1.0f / (float)NROWS);
}

extern "C" void kernel_launch(void* const* d_in, const int* in_sizes, int n_in,
                              void* d_out, int out_size, void* d_ws, size_t ws_size,
                              hipStream_t stream) {
  const float* z = (const float*)d_in[0];
  float* out = (float*)d_out;

  _Float16* zn = (_Float16*)d_ws;                                   // 4 MiB
  float* acc = (float*)((char*)d_ws + (size_t)NROWS * KDIM * sizeof(_Float16));

  infonce_norm_kernel<<<NROWS / 8, 256, 0, stream>>>(z, zn, acc);
  infonce_sim_kernel<<<NROWS / 32, 256, 0, stream>>>(zn, acc);
  infonce_final_kernel<<<1, 32, 0, stream>>>(acc, out);
}
